// FeMaSRNet_14353780703888
// MI455X (gfx1250) — matrix operand order repaired and checked by reading, not compiled, not run on hardware
//
#include <hip/hip_runtime.h>

// ---------------------------------------------------------------------------
// VQ forward for MI455X (gfx1250, wave32, WMMA + TDM).
//   z:        [16, 512, 64, 64] f32   (B, C, H, W), N = B*H*W = 65536 pixels
//   codebook: [2048, 512] f32
//   out:      z_q [16,512,64,64] f32 (33554432 floats) + loss scalar (1 float)
// Distance GEMM in f16 WMMA (f32 accumulate); loss/output from original f32.
// Codebook chunks staged into LDS by the Tensor Data Mover, double-buffered.
// ---------------------------------------------------------------------------

typedef __attribute__((ext_vector_type(16))) _Float16 v16h;
typedef __attribute__((ext_vector_type(8)))  _Float16 v8h;
typedef __attribute__((ext_vector_type(8)))  float    v8f;
typedef __attribute__((ext_vector_type(4)))  unsigned int u32x4;
typedef __attribute__((ext_vector_type(8)))  int          i32x8;
typedef __attribute__((ext_vector_type(4)))  int          i32x4;

union U16 { v16h v; v8h h[2]; };

#define NPIX      65536      // B*H*W
#define CDIM      512
#define NE        2048
#define HWSZ      4096       // H*W
#define ZS        520        // padded LDS row stride in halves (1040 B -> 4-bank lane stride)
#define NCOUT     33554432   // B*C*H*W
#define CHUNK     64         // codes per LDS chunk
#define NCHUNK    (NE / CHUNK)

// ---------------------------------------------------------------------------
// Kernel 1: codebook f32 -> f16, per-code squared norms, zero loss accumulator
// ---------------------------------------------------------------------------
__global__ void vq_prep(const float* __restrict__ cb, _Float16* __restrict__ cb16,
                        float* __restrict__ cbnorm, float* __restrict__ lossacc) {
    __shared__ float red[8];
    const int e = blockIdx.x;
    const int t = threadIdx.x;
    const int base = e * CDIM;
    float v0 = cb[base + t];
    float v1 = cb[base + t + 256];
    cb16[base + t]       = (_Float16)v0;
    cb16[base + t + 256] = (_Float16)v1;
    float s = v0 * v0 + v1 * v1;
    #pragma unroll
    for (int m = 16; m > 0; m >>= 1) s += __shfl_xor(s, m, 32);
    if ((t & 31) == 0) red[t >> 5] = s;
    __syncthreads();
    if (t == 0) {
        float tot = 0.f;
        #pragma unroll
        for (int w = 0; w < 8; ++w) tot += red[w];
        cbnorm[e] = tot;
        if (e == 0) lossacc[0] = 0.f;   // re-zeroed every launch (graph-safe)
    }
}

// ---------------------------------------------------------------------------
// TDM: load one 64 x 512-half codebook chunk into LDS with padded rows.
// D# group0: count=1 | lds_addr[63:32] | global_addr[120:64] | type=2 [127:126]
// D# group1: data_size=2B, pad every 256 DWORDs by 4 DWORDs (512 -> ZS halves),
//            tensor 512 x 64, tile 512 x 64, dim0 stride 512.
// Groups 2/3 unused (2D tensor); 6-arg builtin form (this toolchain).
// ---------------------------------------------------------------------------
__device__ __forceinline__ void tdm_load_chunk(const _Float16* gsrc, unsigned ldsOff) {
    const unsigned long long ga = (unsigned long long)(uintptr_t)gsrc;
    u32x4 g0;
    g0[0] = 1u;                                                   // count=1 user D#
    g0[1] = ldsOff;                                               // lds_addr (bytes)
    g0[2] = (unsigned)ga;                                         // global_addr[31:0]
    g0[3] = ((unsigned)(ga >> 32) & 0x01FFFFFFu) | 0x80000000u;   // addr[56:32] | type=2
    i32x8 g1;
    g1[0] = (1 << 16) | (1 << 20) | (7 << 22) | (3 << 25);  // data_size=2B, pad_en, ivl=256DW, amt=4DW
    g1[1] = (int)(512u << 16);                              // tensor_dim0 = 512 (bits 79:48)
    g1[2] = (int)(64u  << 16);                              // tensor_dim1 = 64  (bits 111:80)
    g1[3] = (int)(512u << 16);                              // tile_dim0   = 512 (bits 127:112)
    g1[4] = 64;                                             // tile_dim1 = 64, tile_dim2 = 0
    g1[5] = 512;                                            // tensor_dim0_stride = 512
    g1[6] = 0;                                              // stride0 hi / stride1 lo
    g1[7] = 0;
    const i32x4 z4 = {0, 0, 0, 0};
    const i32x8 z8 = {0, 0, 0, 0, 0, 0, 0, 0};
    __builtin_amdgcn_tensor_load_to_lds(g0, g1, z4, z4, z8, 0);
}

// ---------------------------------------------------------------------------
// Kernel 2: argmin over codebook via WMMA.
// Block: 256 threads = 8 wave32, 128 pixels per block (16 per wave).
// LDS: z-tile 128 x ZS halves + 2 x (64 x ZS) TDM-staged chunk buffers.
// Distance surrogate: ||e||^2 - 2 * (z . e)   (||z||^2 constant per row).
// ---------------------------------------------------------------------------
__global__ void vq_argmin(const float* __restrict__ z,
                          const _Float16* __restrict__ cb16,
                          const float* __restrict__ cbnorm,
                          unsigned* __restrict__ idxbuf) {
    extern __shared__ char smem[];
    _Float16* zt  = (_Float16*)smem;                              // [128][ZS]
    _Float16* ct0 = (_Float16*)(smem + 128 * ZS * 2);             // [64][ZS]
    _Float16* ct1 = (_Float16*)(smem + (128 + 64) * ZS * 2);      // [64][ZS]
    const unsigned ct0Off = (unsigned)(uintptr_t)(void*)ct0;      // LDS byte offsets
    const unsigned ct1Off = (unsigned)(uintptr_t)(void*)ct1;

    const int t       = threadIdx.x;
    const int lane    = t & 31;
    const int wv      = t >> 5;            // wave id 0..7
    const int ln      = lane & 15;
    const int halfsel = lane >> 4;         // 0: lanes 0-15, 1: lanes 16-31
    const int pix0    = blockIdx.x * 128;
    const size_t bimg = (size_t)(pix0 >> 12);              // image index (128 | 4096)
    const int hw0     = pix0 & (HWSZ - 1);

    // ---- stage z tile: zt[i][c] = z[bimg, c, hw0+i] as f16 (coalesced reads)
    {
        const int i = t & 127;
        const size_t gbase = bimg * (size_t)(CDIM * HWSZ) + (size_t)hw0 + (size_t)i;
        for (int c = (t >> 7); c < CDIM; c += 2)
            zt[i * ZS + c] = (_Float16)z[gbase + (size_t)c * HWSZ];
    }

    float    minv[8];
    unsigned mini[8];
    #pragma unroll
    for (int j = 0; j < 8; ++j) { minv[j] = 3.4e38f; mini[j] = 0u; }

    const int zrow = (wv * 16 + ln) * ZS;
    const int aoff = halfsel << 4;        // A: contiguous 16 K values per lane-half
    const int boff = halfsel << 4;        // B: contiguous 16 K values per lane-half

    auto loadA = [&](U16& a, int k0) {
        const _Float16* p = zt + zrow + k0 + aoff;
        a.h[0] = *(const v8h*)p;
        a.h[1] = *(const v8h*)(p + 8);
    };
    auto loadB = [&](U16& b, const _Float16* ctb, int tt, int k0) {
        const v8h* pb = (const v8h*)(ctb + (tt * 16 + ln) * ZS + k0 + boff);
        b.h[0] = pb[0];
        b.h[1] = pb[1];
    };

    // ---- TDM prologue: chunk 0 -> buffer 0
    if (wv == 0) tdm_load_chunk(cb16, ct0Off);

    for (int n = 0; n < NCHUNK; ++n) {
        __syncthreads();   // all waves done reading buf[(n+1)&1] (iter n-1); safe to overwrite
        if (wv == 0) {
            if (n + 1 < NCHUNK) {
                tdm_load_chunk(cb16 + (size_t)(n + 1) * CHUNK * CDIM,
                               ((n + 1) & 1) ? ct1Off : ct0Off);
                __builtin_amdgcn_s_wait_tensorcnt(1);   // chunk n complete (in-order)
            } else {
                __builtin_amdgcn_s_wait_tensorcnt(0);
            }
        }
        __syncthreads();   // publish buf[n&1] to all waves
        const _Float16* ctb = (n & 1) ? ct1 : ct0;

        v8f acc[4];
        #pragma unroll
        for (int tt = 0; tt < 4; ++tt)
            acc[tt] = (v8f){0.f, 0.f, 0.f, 0.f, 0.f, 0.f, 0.f, 0.f};

        // ---- software-pipelined K loop: next-step loads overlap current WMMAs
        U16 aC, b0C, b1C, b2C, b3C;
        loadA(aC, 0);
        loadB(b0C, ctb, 0, 0); loadB(b1C, ctb, 1, 0);
        loadB(b2C, ctb, 2, 0); loadB(b3C, ctb, 3, 0);
        for (int k0 = 32; k0 < CDIM; k0 += 32) {
            U16 aN, b0N, b1N, b2N, b3N;
            loadA(aN, k0);
            loadB(b0N, ctb, 0, k0); loadB(b1N, ctb, 1, k0);
            loadB(b2N, ctb, 2, k0); loadB(b3N, ctb, 3, k0);
            acc[0] = __builtin_amdgcn_wmma_f32_16x16x32_f16(false, aC.v, false, b0C.v, (short)0, acc[0], false, false);
            acc[1] = __builtin_amdgcn_wmma_f32_16x16x32_f16(false, aC.v, false, b1C.v, (short)0, acc[1], false, false);
            acc[2] = __builtin_amdgcn_wmma_f32_16x16x32_f16(false, aC.v, false, b2C.v, (short)0, acc[2], false, false);
            acc[3] = __builtin_amdgcn_wmma_f32_16x16x32_f16(false, aC.v, false, b3C.v, (short)0, acc[3], false, false);
            aC = aN; b0C = b0N; b1C = b1N; b2C = b2N; b3C = b3N;
        }
        acc[0] = __builtin_amdgcn_wmma_f32_16x16x32_f16(false, aC.v, false, b0C.v, (short)0, acc[0], false, false);
        acc[1] = __builtin_amdgcn_wmma_f32_16x16x32_f16(false, aC.v, false, b1C.v, (short)0, acc[1], false, false);
        acc[2] = __builtin_amdgcn_wmma_f32_16x16x32_f16(false, aC.v, false, b2C.v, (short)0, acc[2], false, false);
        acc[3] = __builtin_amdgcn_wmma_f32_16x16x32_f16(false, aC.v, false, b3C.v, (short)0, acc[3], false, false);

        // ---- fold chunk into running per-lane argmin
        const int e0 = n * CHUNK;
        #pragma unroll
        for (int tt = 0; tt < 4; ++tt) {
            const unsigned e = (unsigned)(e0 + tt * 16 + ln);
            const float en = cbnorm[e];
            #pragma unroll
            for (int j = 0; j < 8; ++j) {
                const float dist = en - 2.0f * acc[tt][j];
                if (dist < minv[j]) { minv[j] = dist; mini[j] = e; }
            }
        }
    }

    // ---- cross-lane reduce (codes are striped e % 16 == ln within each half)
    #pragma unroll
    for (int j = 0; j < 8; ++j) {
        float v = minv[j];
        unsigned ii = mini[j];
        #pragma unroll
        for (int m = 1; m < 16; m <<= 1) {
            const float    ov = __shfl_xor(v, m, 32);
            const unsigned oi = (unsigned)__shfl_xor((int)ii, m, 32);
            if (ov < v || (ov == v && oi < ii)) { v = ov; ii = oi; }
        }
        if (ln == 0)
            idxbuf[pix0 + wv * 16 + halfsel * 8 + j] = ii;
    }
}

// ---------------------------------------------------------------------------
// Kernel 3: gather z_q = codebook[idx], transpose to [B,C,H,W] via LDS,
// accumulate sum((z_q - z)^2) in fp32.
// ---------------------------------------------------------------------------
#define PADF 513
__global__ void vq_gather(const float* __restrict__ z,
                          const float* __restrict__ cb,
                          const unsigned* __restrict__ idxbuf,
                          float* __restrict__ out,
                          float* __restrict__ lossacc) {
    extern __shared__ char smem[];
    float* tile = (float*)smem;                                   // [64][PADF]
    int*   sidx = (int*)(smem + 64 * PADF * 4);                   // [64]
    float* part = (float*)(smem + 64 * PADF * 4 + 64 * 4);        // [8]

    const int t    = threadIdx.x;
    const int pix0 = blockIdx.x * 64;
    if (t < 64) sidx[t] = (int)idxbuf[pix0 + t];
    __syncthreads();

    // ---- gather codebook rows (contiguous per row) into LDS, pixel-major
    {
        const int pr = t >> 2;      // pixel 0..63
        const int q  = t & 3;
        const float4* src = (const float4*)(cb + (size_t)sidx[pr] * CDIM);
        float* drow = tile + pr * PADF;
        for (int u = q; u < CDIM / 4; u += 4) {
            const float4 v = src[u];
            drow[u * 4 + 0] = v.x;
            drow[u * 4 + 1] = v.y;
            drow[u * 4 + 2] = v.z;
            drow[u * 4 + 3] = v.w;
        }
    }
    __syncthreads();

    // ---- write coalesced in [B,C,H,W]; fuse loss accumulation
    const int i  = t & 63;          // pixel within tile
    const int cs = t >> 6;          // 0..3
    const size_t bimg  = (size_t)(pix0 >> 12);
    const int    hw0   = pix0 & (HWSZ - 1);
    const size_t gbase = bimg * (size_t)(CDIM * HWSZ) + (size_t)hw0 + (size_t)i;
    float lsum = 0.f;
    for (int c = cs; c < CDIM; c += 4) {
        const float  v  = tile[i * PADF + c];
        const size_t ga = gbase + (size_t)c * HWSZ;
        const float  d  = v - z[ga];
        lsum += d * d;
        out[ga] = v;
    }
    #pragma unroll
    for (int m = 16; m > 0; m >>= 1) lsum += __shfl_xor(lsum, m, 32);
    if ((t & 31) == 0) part[t >> 5] = lsum;
    __syncthreads();
    if (t == 0) {
        float s = 0.f;
        #pragma unroll
        for (int w = 0; w < 8; ++w) s += part[w];
        atomicAdd(lossacc, s);
    }
}

// ---------------------------------------------------------------------------
// Kernel 4: loss = (1 + beta) * sum / numel,  beta = 0.25
// ---------------------------------------------------------------------------
__global__ void vq_finalize(const float* __restrict__ lossacc, float* __restrict__ out) {
    out[NCOUT] = lossacc[0] * 1.25f / (float)NCOUT;
}

// ---------------------------------------------------------------------------
extern "C" void kernel_launch(void* const* d_in, const int* in_sizes, int n_in,
                              void* d_out, int out_size, void* d_ws, size_t ws_size,
                              hipStream_t stream) {
    const float* z  = (const float*)d_in[0];
    const float* cb = (const float*)d_in[1];
    float* out = (float*)d_out;

    // workspace layout (~2.37 MB)
    char* ws = (char*)d_ws;
    _Float16* cb16    = (_Float16*)ws;                               // 2 MB
    float*    cbnorm  = (float*)(ws + 2u * 1024u * 1024u);           // 8 KB
    unsigned* idxbuf  = (unsigned*)(ws + 2u * 1024u * 1024u + 8192u);// 256 KB
    float*    lossacc = (float*)(ws + 2u * 1024u * 1024u + 8192u + 262144u);

    vq_prep<<<NE, 256, 0, stream>>>(cb, cb16, cbnorm, lossacc);

    const size_t lds1 = (size_t)(128 + 64 + 64) * ZS * 2;            // 266240 B
    (void)hipFuncSetAttribute(reinterpret_cast<const void*>(vq_argmin),
                              hipFuncAttributeMaxDynamicSharedMemorySize, (int)lds1);
    vq_argmin<<<NPIX / 128, 256, lds1, stream>>>(z, cb16, cbnorm, idxbuf);

    const size_t lds2 = (size_t)64 * PADF * 4 + 64 * 4 + 8 * 4;      // 131616 B
    (void)hipFuncSetAttribute(reinterpret_cast<const void*>(vq_gather),
                              hipFuncAttributeMaxDynamicSharedMemorySize, (int)lds2);
    vq_gather<<<NPIX / 64, 256, lds2, stream>>>(z, cb, idxbuf, out, lossacc);

    vq_finalize<<<1, 1, 0, stream>>>(lossacc, out);
}
